// Classifier_31507880083763
// MI455X (gfx1250) — matrix operand order repaired and checked by reading, not compiled, not run on hardware
//
#include <hip/hip_runtime.h>
#include <hip/hip_bf16.h>

#define NN   50000
#define EE   800000
#define DIN_ 128
#define HH   256
#define GG   256
#define CC   16
#define BN_S 0.9999950000375f   // 1/sqrt(1 + 1e-5)

typedef __attribute__((ext_vector_type(16))) _Float16 v16h;
typedef __attribute__((ext_vector_type(8)))  _Float16 v8h;
typedef __attribute__((ext_vector_type(4)))  _Float16 v4h;
typedef __attribute__((ext_vector_type(8)))  float    v8f;

// ---------------- weight prep: convert fp32 [fin x fout] -> fp16 [fout x fin] ----------------
__global__ void wprep_kernel(const float* __restrict__ W, _Float16* __restrict__ Wt,
                             int fin, int fout) {
    int i = blockIdx.x * blockDim.x + threadIdx.x;
    if (i >= fin * fout) return;
    int o = i / fin, k = i - o * fin;                 // coalesced writes over k
    Wt[(size_t)o * fin + k] = (_Float16)W[(size_t)k * fout + o];
}

// ---------------- edge scatter-add: agg[dst] += x[src]  ----------------
__global__ void edge_f32_kernel(const float* __restrict__ x, const int* __restrict__ esrc,
                                const int* __restrict__ edst, float* __restrict__ agg) {
    int t = blockIdx.x * blockDim.x + threadIdx.x;    // EE * (DIN_/4) threads
    if (t >= EE * (DIN_ / 4)) return;
    int e = t >> 5, c = (t & 31) * 4;
    int s = esrc[e], d = edst[e];
    float4 v = *(const float4*)(x + (size_t)s * DIN_ + c);
    float* p = agg + (size_t)d * DIN_ + c;
    atomicAdd(p + 0, v.x); atomicAdd(p + 1, v.y);
    atomicAdd(p + 2, v.z); atomicAdd(p + 3, v.w);
}

__global__ void edge_f16_kernel(const _Float16* __restrict__ x, const int* __restrict__ esrc,
                                const int* __restrict__ edst, float* __restrict__ agg) {
    int t = blockIdx.x * blockDim.x + threadIdx.x;    // EE * (HH/4) threads
    if (t >= EE * (HH / 4)) return;
    int e = t >> 6, c = (t & 63) * 4;
    int s = esrc[e], d = edst[e];
    v4h v = *(const v4h*)(x + (size_t)s * HH + c);
    float* p = agg + (size_t)d * HH + c;
    atomicAdd(p + 0, (float)v[0]); atomicAdd(p + 1, (float)v[1]);
    atomicAdd(p + 2, (float)v[2]); atomicAdd(p + 3, (float)v[3]);
}

// ---------------- asum = (x + agg) converted to fp16 ----------------
__global__ void asum_f32_kernel(const float* __restrict__ x, const float* __restrict__ agg,
                                _Float16* __restrict__ o, int n) {
    int i = blockIdx.x * blockDim.x + threadIdx.x;
    if (i < n) o[i] = (_Float16)(x[i] + agg[i]);
}
__global__ void asum_f16_kernel(const _Float16* __restrict__ x, const float* __restrict__ agg,
                                _Float16* __restrict__ o, int n) {
    int i = blockIdx.x * blockDim.x + threadIdx.x;
    if (i < n) o[i] = (_Float16)((float)x[i] + agg[i]);
}

// ---------------- WMMA GEMM: out[N x 256] = A[N x K] @ Bt^T, fused epilogues ----------------
// MODE 0: y = relu(gamma*((acc+bias)*BN_S)+beta)      -> hbuf (fp16)
// MODE 1: y = relu(acc+bias) -> x_l (fp16) + pool atomicMax (int-bit trick, y>=0)
template <int MODE>
__global__ __launch_bounds__(256) void gin_gemm_kernel(
    const _Float16* __restrict__ A, const _Float16* __restrict__ Bt,
    const float* __restrict__ bias, const float* __restrict__ gamma,
    const float* __restrict__ beta, _Float16* __restrict__ out,
    const int* __restrict__ batch, int* __restrict__ pool, int sec,
    int nrows, int K) {
    const int lane  = threadIdx.x & 31;
    const int wave  = threadIdx.x >> 5;
    const int mBase = blockIdx.x * 32 + (wave >> 2) * 16;
    const int nBase = (wave & 3) * 64;
    const int r     = lane & 15;
    const int hi    = lane >> 4;

    int row  = mBase + r;
    int rowc = row < nrows ? row : nrows - 1;          // clamp: keep EXEC all-ones for WMMA
    const int ncol = nBase + r;

    // A frag (16x32 f16): lane=row, 16 contiguous K at hi*16                (two 16B loads)
    const _Float16* Ap  = A  + (size_t)rowc * K + hi * 16;
    // B frag (32x16 f16): lane=col, 16 contiguous K at hi*16               (one 32B load)
    const _Float16* Bp0 = Bt + (size_t)(ncol +  0) * K + hi * 16;
    const _Float16* Bp1 = Bt + (size_t)(ncol + 16) * K + hi * 16;
    const _Float16* Bp2 = Bt + (size_t)(ncol + 32) * K + hi * 16;
    const _Float16* Bp3 = Bt + (size_t)(ncol + 48) * K + hi * 16;

    v8f acc0 = {}, acc1 = {}, acc2 = {}, acc3 = {};
    for (int ko = 0; ko < K; ko += 32) {
        v8h a0 = *(const v8h*)(Ap + ko);
        v8h a1 = *(const v8h*)(Ap + ko + 8);
        v16h a = __builtin_shufflevector(a0, a1, 0,1,2,3,4,5,6,7,8,9,10,11,12,13,14,15);
        v16h b0 = *(const v16h*)(Bp0 + ko);
        v16h b1 = *(const v16h*)(Bp1 + ko);
        v16h b2 = *(const v16h*)(Bp2 + ko);
        v16h b3 = *(const v16h*)(Bp3 + ko);
        acc0 = __builtin_amdgcn_wmma_f32_16x16x32_f16(false, a, false, b0, (short)0, acc0, false, false);
        acc1 = __builtin_amdgcn_wmma_f32_16x16x32_f16(false, a, false, b1, (short)0, acc1, false, false);
        acc2 = __builtin_amdgcn_wmma_f32_16x16x32_f16(false, a, false, b2, (short)0, acc2, false, false);
        acc3 = __builtin_amdgcn_wmma_f32_16x16x32_f16(false, a, false, b3, (short)0, acc3, false, false);
    }

    v8f accs[4] = {acc0, acc1, acc2, acc3};
    const int mOff = mBase + hi * 8;                   // C/D layout: VGPR i -> row hi*8+i
#pragma unroll
    for (int t = 0; t < 4; ++t) {
        const int n  = nBase + t * 16 + r;
        const float bv = bias[n];
        float gm = 0.f, bt = 0.f;
        if (MODE == 0) { gm = gamma[n]; bt = beta[n]; }
#pragma unroll
        for (int i = 0; i < 8; ++i) {
            const int m = mOff + i;
            if (m < nrows) {
                float y;
                if (MODE == 0) y = gm * ((accs[t][i] + bv) * BN_S) + bt;
                else           y = accs[t][i] + bv;
                y = fmaxf(y, 0.0f);
                out[(size_t)m * HH + n] = (_Float16)y;
                if (MODE == 1)
                    atomicMax(pool + (size_t)batch[m] * (3 * HH) + sec + n,
                              __float_as_int(y));
            }
        }
    }
}

// ---------------- classifier: out[G x 16] = pool[G x 768] @ Wc + bc ----------------
__global__ void cls_kernel(const int* __restrict__ pool, const float* __restrict__ Wc,
                           const float* __restrict__ bc, float* __restrict__ out) {
    int idx = blockIdx.x * blockDim.x + threadIdx.x;   // GG*CC = 4096
    if (idx >= GG * CC) return;
    int g = idx >> 4, c = idx & 15;
    const int* pr = pool + (size_t)g * (3 * HH);
    float s = bc[c];
    for (int k = 0; k < 3 * HH; ++k)
        s += __int_as_float(pr[k]) * Wc[(size_t)k * CC + c];
    out[idx] = s;
}

// ---------------- launch ----------------
extern "C" void kernel_launch(void* const* d_in, const int* in_sizes, int n_in,
                              void* d_out, int out_size, void* d_ws, size_t ws_size,
                              hipStream_t stream) {
    const float* x     = (const float*)d_in[0];
    const int*   eidx  = (const int*)d_in[1];
    const int*   esrc  = eidx;
    const int*   edst  = eidx + EE;
    const int*   batch = (const int*)d_in[2];
    const float* W[6]  = { (const float*)d_in[3],  (const float*)d_in[7],
                           (const float*)d_in[9],  (const float*)d_in[13],
                           (const float*)d_in[15], (const float*)d_in[19] };
    const float* b11 = (const float*)d_in[4],  *g1 = (const float*)d_in[5],  *be1 = (const float*)d_in[6],  *b12 = (const float*)d_in[8];
    const float* b21 = (const float*)d_in[10], *g2 = (const float*)d_in[11], *be2 = (const float*)d_in[12], *b22 = (const float*)d_in[14];
    const float* b31 = (const float*)d_in[16], *g3 = (const float*)d_in[17], *be3 = (const float*)d_in[18], *b32 = (const float*)d_in[20];
    const float* Wc = (const float*)d_in[21];
    const float* bc = (const float*)d_in[22];

    char* ws = (char*)d_ws;
    size_t off = 0;
    float*    agg  = (float*)(ws + off);    off += (size_t)NN * HH * 4;
    _Float16* asum = (_Float16*)(ws + off); off += (size_t)NN * HH * 2;
    _Float16* hbuf = (_Float16*)(ws + off); off += (size_t)NN * HH * 2;
    _Float16* x1h  = (_Float16*)(ws + off); off += (size_t)NN * HH * 2;
    _Float16* x2h  = (_Float16*)(ws + off); off += (size_t)NN * HH * 2;
    _Float16* x3h  = (_Float16*)(ws + off); off += (size_t)NN * HH * 2;
    int*      pool = (int*)(ws + off);      off += (size_t)GG * 3 * HH * 4;
    _Float16* wt[6];
    int fins[6] = { DIN_, HH, HH, HH, HH, HH };
    for (int i = 0; i < 6; ++i) { wt[i] = (_Float16*)(ws + off); off += (size_t)HH * fins[i] * 2; }

    // weight transpose+convert (tiny)
    for (int i = 0; i < 6; ++i) {
        int cnt = fins[i] * HH;
        wprep_kernel<<<(cnt + 255) / 256, 256, 0, stream>>>(W[i], wt[i], fins[i], HH);
    }
    hipMemsetAsync(pool, 0, (size_t)GG * 3 * HH * 4, stream);   // relu -> max >= 0

    const int gemm_grid = (NN + 31) / 32;

    // ---- layer 1 (K = 128, fp32 input) ----
    hipMemsetAsync(agg, 0, (size_t)NN * DIN_ * 4, stream);
    edge_f32_kernel<<<(EE * (DIN_ / 4) + 255) / 256, 256, 0, stream>>>(x, esrc, edst, agg);
    asum_f32_kernel<<<(NN * DIN_ + 255) / 256, 256, 0, stream>>>(x, agg, asum, NN * DIN_);
    gin_gemm_kernel<0><<<gemm_grid, 256, 0, stream>>>(asum, wt[0], b11, g1, be1, hbuf,
                                                      nullptr, nullptr, 0, NN, DIN_);
    gin_gemm_kernel<1><<<gemm_grid, 256, 0, stream>>>(hbuf, wt[1], b12, nullptr, nullptr, x1h,
                                                      batch, pool, 0, NN, HH);
    // ---- layer 2 (K = 256, fp16 input) ----
    hipMemsetAsync(agg, 0, (size_t)NN * HH * 4, stream);
    edge_f16_kernel<<<(EE * (HH / 4) + 255) / 256, 256, 0, stream>>>(x1h, esrc, edst, agg);
    asum_f16_kernel<<<(NN * HH + 255) / 256, 256, 0, stream>>>(x1h, agg, asum, NN * HH);
    gin_gemm_kernel<0><<<gemm_grid, 256, 0, stream>>>(asum, wt[2], b21, g2, be2, hbuf,
                                                      nullptr, nullptr, 0, NN, HH);
    gin_gemm_kernel<1><<<gemm_grid, 256, 0, stream>>>(hbuf, wt[3], b22, nullptr, nullptr, x2h,
                                                      batch, pool, HH, NN, HH);
    // ---- layer 3 ----
    hipMemsetAsync(agg, 0, (size_t)NN * HH * 4, stream);
    edge_f16_kernel<<<(EE * (HH / 4) + 255) / 256, 256, 0, stream>>>(x2h, esrc, edst, agg);
    asum_f16_kernel<<<(NN * HH + 255) / 256, 256, 0, stream>>>(x2h, agg, asum, NN * HH);
    gin_gemm_kernel<0><<<gemm_grid, 256, 0, stream>>>(asum, wt[4], b31, g3, be3, hbuf,
                                                      nullptr, nullptr, 0, NN, HH);
    gin_gemm_kernel<1><<<gemm_grid, 256, 0, stream>>>(hbuf, wt[5], b32, nullptr, nullptr, x3h,
                                                      batch, pool, 2 * HH, NN, HH);

    // ---- classifier ----
    cls_kernel<<<(GG * CC + 255) / 256, 256, 0, stream>>>(pool, Wc, bc, (float*)d_out);
}
